// EfficientSelfAttention_64055142253130
// MI455X (gfx1250) — matrix operand order repaired and checked by reading, not compiled, not run on hardware
//
#include <hip/hip_runtime.h>
#include <cstdint>
#include <cstddef>

// ---------------------------------------------------------------------------
// EfficientSelfAttention for MI455X (gfx1250), wave32, WMMA bf16 path.
// B=4, N=4096, D=1024, H=16, d_head=64.
// All fragments are loaded with contiguous global_load_b128 (no gathers):
// weights / V / ctx are kept in transposed (column-native) layouts.
// ---------------------------------------------------------------------------

typedef __attribute__((ext_vector_type(16))) __bf16 v16bf;
typedef __attribute__((ext_vector_type(8)))  float  v8f;

union BF16Frag {
  v16bf v;
  unsigned short u[16];
  uint4 q[2];
};

__device__ __forceinline__ unsigned short f32_to_bf16(float f) {
  unsigned u = __float_as_uint(f);
  u += 0x7FFFu + ((u >> 16) & 1u);   // round-to-nearest-even
  return (unsigned short)(u >> 16);
}
__device__ __forceinline__ float bf16_to_f32(unsigned short h) {
  return __uint_as_float(((unsigned)h) << 16);
}
__device__ __forceinline__ unsigned pack_bf16x2(float a, float b) {
  return (unsigned)f32_to_bf16(a) | ((unsigned)f32_to_bf16(b) << 16);
}

#define SM_SCALE 0.5946035575013605f   /* (64^0.5)^(-0.25) = 8^(-0.25) */

// ---------------------------------------------------------------------------
// 0a) f32 -> bf16 convert (x)
// ---------------------------------------------------------------------------
__global__ void cvt_f32_bf16(const float* __restrict__ src,
                             unsigned short* __restrict__ dst, int n) {
  int i = blockIdx.x * blockDim.x + threadIdx.x;
  if (i < n) dst[i] = f32_to_bf16(src[i]);
}

// ---------------------------------------------------------------------------
// 0b) tiled transpose + convert: Wt[n][k] = (bf16) W[k][n]   (D x D)
// ---------------------------------------------------------------------------
__global__ void transpose_cvt(const float* __restrict__ src,
                              unsigned short* __restrict__ dst, int Dm) {
  __shared__ float tile[32][33];
  const int bx = blockIdx.x * 32, by = blockIdx.y * 32;
  const int tx = threadIdx.x, ty = threadIdx.y;   // (32, 8)
  #pragma unroll
  for (int r = 0; r < 32; r += 8)
    tile[ty + r][tx] = src[(size_t)(by + ty + r) * Dm + bx + tx];
  __syncthreads();
  #pragma unroll
  for (int r = 0; r < 32; r += 8)
    dst[(size_t)(bx + ty + r) * Dm + by + tx] = f32_to_bf16(tile[tx][ty + r]);
}

// ---------------------------------------------------------------------------
// 1) bf16 WMMA GEMM:  C(MxN) = A(MxK) @ Bt^T + bias,  Bt is N x K (row = col of B)
//    Block = 256 threads (8 waves). Each wave: one 16-row tile x four 16-col
//    tiles (16x64 of C), so the A fragment is reused across 4 wmmas.
//    Every fragment = two contiguous 16B global loads per lane.
//    MODE 0: f32 row-major store. MODE 1: bf16 row-major. MODE 2: bf16
//    transposed store Cout[col*M + row] (packed uint4 stores).
// ---------------------------------------------------------------------------
template<int MODE>
__global__ void gemm_wmma_bf16(const unsigned short* __restrict__ A,
                               const unsigned short* __restrict__ Bt,
                               const float* __restrict__ bias,
                               void* __restrict__ Cout,
                               int M, int N, int K) {
  const int wave = threadIdx.x >> 5, lane = threadIdx.x & 31;
  const int lm = lane & 15, lh = lane >> 4;
  const int cbase = blockIdx.x * 64;
  const int rbase = blockIdx.y * 128;
  const int row = rbase + wave * 16 + lm;

  const unsigned short* Arow = A + (size_t)row * K;
  const unsigned short* Bc0 = Bt + (size_t)(cbase + 0 * 16 + lm) * K;
  const unsigned short* Bc1 = Bt + (size_t)(cbase + 1 * 16 + lm) * K;
  const unsigned short* Bc2 = Bt + (size_t)(cbase + 2 * 16 + lm) * K;
  const unsigned short* Bc3 = Bt + (size_t)(cbase + 3 * 16 + lm) * K;

  v8f acc0 = {0.f,0.f,0.f,0.f,0.f,0.f,0.f,0.f};
  v8f acc1 = acc0, acc2 = acc0, acc3 = acc0;

  #pragma unroll 2
  for (int kb = 0; kb < K; kb += 32) {
    __builtin_prefetch(Arow + kb + 256, 0, 1);
    BF16Frag a;   // A 16x32: lane-half lh holds K = lh*16 + j
    a.q[0] = *(const uint4*)(Arow + kb + lh * 16);
    a.q[1] = *(const uint4*)(Arow + kb + lh * 16 + 8);
    BF16Frag b0, b1, b2, b3;   // B 32x16: lane-half lh holds K = lh*16 + j
    b0.q[0] = *(const uint4*)(Bc0 + kb + lh * 16);
    b0.q[1] = *(const uint4*)(Bc0 + kb + lh * 16 + 8);
    b1.q[0] = *(const uint4*)(Bc1 + kb + lh * 16);
    b1.q[1] = *(const uint4*)(Bc1 + kb + lh * 16 + 8);
    b2.q[0] = *(const uint4*)(Bc2 + kb + lh * 16);
    b2.q[1] = *(const uint4*)(Bc2 + kb + lh * 16 + 8);
    b3.q[0] = *(const uint4*)(Bc3 + kb + lh * 16);
    b3.q[1] = *(const uint4*)(Bc3 + kb + lh * 16 + 8);
    acc0 = __builtin_amdgcn_wmma_f32_16x16x32_bf16(false, a.v, false, b0.v, (short)0, acc0, false, false);
    acc1 = __builtin_amdgcn_wmma_f32_16x16x32_bf16(false, a.v, false, b1.v, (short)0, acc1, false, false);
    acc2 = __builtin_amdgcn_wmma_f32_16x16x32_bf16(false, a.v, false, b2.v, (short)0, acc2, false, false);
    acc3 = __builtin_amdgcn_wmma_f32_16x16x32_bf16(false, a.v, false, b3.v, (short)0, acc3, false, false);
  }

  v8f accs[4] = {acc0, acc1, acc2, acc3};
  #pragma unroll
  for (int t = 0; t < 4; ++t) {
    const int col = cbase + t * 16 + lm;
    const float bv = bias ? bias[col] : 0.f;
    if (MODE == 2) {
      // lane's 8 results are 8 consecutive rows at fixed col -> one uint4
      float v0 = accs[t][0] + bv, v1 = accs[t][1] + bv, v2 = accs[t][2] + bv, v3 = accs[t][3] + bv;
      float v4 = accs[t][4] + bv, v5 = accs[t][5] + bv, v6 = accs[t][6] + bv, v7 = accs[t][7] + bv;
      uint4 p = { pack_bf16x2(v0, v1), pack_bf16x2(v2, v3),
                  pack_bf16x2(v4, v5), pack_bf16x2(v6, v7) };
      *(uint4*)((unsigned short*)Cout + (size_t)col * M + rbase + wave * 16 + lh * 8) = p;
    } else {
      #pragma unroll
      for (int r = 0; r < 8; ++r) {
        const int rr = rbase + wave * 16 + lh * 8 + r;   // C rows: half*8 + r
        const float v = accs[t][r] + bv;
        if (MODE == 1)
          ((unsigned short*)Cout)[(size_t)rr * N + col] = f32_to_bf16(v);
        else
          ((float*)Cout)[(size_t)rr * N + col] = v;
      }
    }
  }
}

// ---------------------------------------------------------------------------
// 2) softmax over each 64-element head segment, * s; writes row-major Qs
//    and transposed QsT[col][grow] (col = h*64+d, grow = b*4096+n)
// ---------------------------------------------------------------------------
__global__ void softmax_head_rows(const unsigned short* __restrict__ Qpre,
                                  unsigned short* __restrict__ Qs,
                                  unsigned short* __restrict__ QsT, int total) {
  const int idx = blockIdx.x * blockDim.x + threadIdx.x;
  if (idx >= total) return;
  const int grow = idx >> 4, h = idx & 15;
  const size_t base = (size_t)grow * 1024 + (size_t)h * 64;
  float v[64];
  float mx = -1e30f;
  #pragma unroll
  for (int i = 0; i < 64; ++i) {
    v[i] = bf16_to_f32(Qpre[base + i]);
    mx = fmaxf(mx, v[i]);
  }
  float sum = 0.f;
  #pragma unroll
  for (int i = 0; i < 64; ++i) { v[i] = __expf(v[i] - mx); sum += v[i]; }
  const float inv = SM_SCALE * __builtin_amdgcn_rcpf(sum);
  #pragma unroll
  for (int i = 0; i < 64; ++i) {
    const unsigned short o = f32_to_bf16(v[i] * inv);
    Qs[base + i] = o;
    QsT[(size_t)(h * 64 + i) * 16384 + grow] = o;
  }
}

// ---------------------------------------------------------------------------
// 3) colsum[c] = sum_n exp(qs[n, c])  via QsT (contiguous b128 reads)
//    one block (256 thr) per column; LDS tree reduction
// ---------------------------------------------------------------------------
__global__ void colsum_exp_kernel(const unsigned short* __restrict__ QsT,
                                  float* __restrict__ colsum) {
  __shared__ float red[256];
  const int c = blockIdx.x;                 // 0..4095
  const int b = c >> 10, rem = c & 1023;    // rem = h*64 + d
  const unsigned short* p = QsT + (size_t)rem * 16384 + (size_t)b * 4096;
  const int t = threadIdx.x;
  float s = 0.f;
  #pragma unroll
  for (int it = 0; it < 2; ++it) {
    uint4 d = *(const uint4*)(p + (t + it * 256) * 8);
    const unsigned* w = (const unsigned*)&d;
    #pragma unroll
    for (int i = 0; i < 4; ++i) {
      s += __expf(bf16_to_f32((unsigned short)(w[i] & 0xFFFFu)));
      s += __expf(bf16_to_f32((unsigned short)(w[i] >> 16)));
    }
  }
  red[t] = s;
  __syncthreads();
  for (int off = 128; off; off >>= 1) {
    if (t < off) red[t] += red[t + off];
    __syncthreads();
  }
  if (t == 0) colsum[c] = red[0];
}

// ---------------------------------------------------------------------------
// 4) ctxT = (s/colsum)*exp(qs)^T @ V per head, stored transposed:
//    ctxT[bh*4096 + e*64 + d].  A = exp(QsT rows) on the fly, B = Vt rows.
//    One block per (b,h), 16 waves, each a 16x16 tile; K = N = 4096.
// ---------------------------------------------------------------------------
__device__ __forceinline__ void exp8_to_frag(uint4 d, unsigned short* out) {
  const unsigned* w = (const unsigned*)&d;
  #pragma unroll
  for (int i = 0; i < 4; ++i) {
    const float f0 = __expf(bf16_to_f32((unsigned short)(w[i] & 0xFFFFu)));
    const float f1 = __expf(bf16_to_f32((unsigned short)(w[i] >> 16)));
    out[2 * i]     = f32_to_bf16(f0);
    out[2 * i + 1] = f32_to_bf16(f1);
  }
}

__global__ void ctx_wmma_kernel(const unsigned short* __restrict__ QsT,
                                const unsigned short* __restrict__ Vt,
                                const float* __restrict__ colsum,
                                unsigned short* __restrict__ ctxT) {
  const int bh = blockIdx.x;               // 0..63
  const int b = bh >> 4, h = bh & 15;
  const int wave = threadIdx.x >> 5, lane = threadIdx.x & 31;
  const int tm = wave >> 2, tn = wave & 3;
  const int lm = lane & 15, lh = lane >> 4;
  const int d = tm * 16 + lm;              // A row (d index)
  const int e = tn * 16 + lm;              // B col (e index)
  const unsigned short* Aarow = QsT + (size_t)(h * 64 + d) * 16384 + (size_t)b * 4096;
  const unsigned short* Brow  = Vt  + (size_t)(h * 64 + e) * 16384 + (size_t)b * 4096;

  v8f acc = {0.f,0.f,0.f,0.f,0.f,0.f,0.f,0.f};

  for (int kb = 0; kb < 4096; kb += 32) {
    BF16Frag a, bf;
    const uint4 a0 = *(const uint4*)(Aarow + kb + lh * 16);
    const uint4 a1 = *(const uint4*)(Aarow + kb + lh * 16 + 8);
    exp8_to_frag(a0, &a.u[0]);
    exp8_to_frag(a1, &a.u[8]);
    bf.q[0] = *(const uint4*)(Brow + kb + lh * 16);
    bf.q[1] = *(const uint4*)(Brow + kb + lh * 16 + 8);
    acc = __builtin_amdgcn_wmma_f32_16x16x32_bf16(
        false, a.v, false, bf.v, (short)0, acc, false, false);
  }

  // lane holds rows d' = tm*16 + lh*8 + r (8 consecutive) at fixed col e
  float v[8];
  #pragma unroll
  for (int r = 0; r < 8; ++r) {
    const int dd = tm * 16 + lh * 8 + r;
    v[r] = acc[r] * SM_SCALE * __builtin_amdgcn_rcpf(colsum[bh * 64 + dd]);
  }
  uint4 p = { pack_bf16x2(v[0], v[1]), pack_bf16x2(v[2], v[3]),
              pack_bf16x2(v[4], v[5]), pack_bf16x2(v[6], v[7]) };
  *(uint4*)(ctxT + (size_t)bh * 4096 + (size_t)e * 64 + tm * 16 + lh * 8) = p;
}

// ---------------------------------------------------------------------------
// 5) attn = qs @ ctx per head, merged-head (B*N, D) bf16 layout.
//    A from row-major Qs (2 b128), B from ctxT rows (2 b128).
// ---------------------------------------------------------------------------
__global__ void attn_wmma_kernel(const unsigned short* __restrict__ Qs,
                                 const unsigned short* __restrict__ ctxT,
                                 unsigned short* __restrict__ attn) {
  const int bh = blockIdx.y;
  const int b = bh >> 4, h = bh & 15;
  const int rbase = blockIdx.x * 32;
  const int wave = threadIdx.x >> 5, lane = threadIdx.x & 31;
  const int tm = wave >> 2, tn = wave & 3;  // tm 0..1, tn 0..3
  const int lm = lane & 15, lh = lane >> 4;
  const int row = rbase + tm * 16 + lm;     // sequence index n
  const unsigned short* Arow = Qs + ((size_t)(b * 4096 + row)) * 1024 + h * 64;
  const unsigned short* Brow = ctxT + (size_t)bh * 4096 + (size_t)(tn * 16 + lm) * 64;

  v8f acc = {0.f,0.f,0.f,0.f,0.f,0.f,0.f,0.f};

  #pragma unroll
  for (int kb = 0; kb < 64; kb += 32) {
    BF16Frag a, bf;
    a.q[0] = *(const uint4*)(Arow + kb + lh * 16);
    a.q[1] = *(const uint4*)(Arow + kb + lh * 16 + 8);
    bf.q[0] = *(const uint4*)(Brow + kb + lh * 16);
    bf.q[1] = *(const uint4*)(Brow + kb + lh * 16 + 8);
    acc = __builtin_amdgcn_wmma_f32_16x16x32_bf16(
        false, a.v, false, bf.v, (short)0, acc, false, false);
  }

  #pragma unroll
  for (int r = 0; r < 8; ++r) {
    const int rr = rbase + tm * 16 + lh * 8 + r;
    attn[((size_t)(b * 4096 + rr)) * 1024 + h * 64 + tn * 16 + lm] =
        f32_to_bf16(acc[r]);
  }
}

// ---------------------------------------------------------------------------
// host-side launch
// ---------------------------------------------------------------------------
extern "C" void kernel_launch(void* const* d_in, const int* in_sizes, int n_in,
                              void* d_out, int out_size, void* d_ws, size_t ws_size,
                              hipStream_t stream) {
  (void)in_sizes; (void)n_in; (void)out_size; (void)ws_size;

  const float* x  = (const float*)d_in[0];
  const float* Wq = (const float*)d_in[1];
  const float* bq = (const float*)d_in[2];
  // d_in[3]=Wk, d_in[4]=bk : dead code in the reference, intentionally skipped
  const float* Wv = (const float*)d_in[5];
  const float* bv = (const float*)d_in[6];
  const float* Wo = (const float*)d_in[7];
  const float* bo = (const float*)d_in[8];
  float* out = (float*)d_out;

  const int Nseq = 4096, D = 1024;
  const int M = 4 * Nseq;                  // 16384
  const size_t Sx = (size_t)M * D;         // 16,777,216 elements

  // workspace layout (bytes, 16B-aligned by construction)
  char* ws = (char*)d_ws;
  unsigned short* Xb   = (unsigned short*)(ws);                 // 33.5 MB
  unsigned short* WqT  = (unsigned short*)(ws + 33554432);      //  2.1 MB
  unsigned short* WvT  = (unsigned short*)(ws + 35651584);      //  2.1 MB
  unsigned short* WoT  = (unsigned short*)(ws + 37748736);      //  2.1 MB
  unsigned short* Qpre = (unsigned short*)(ws + 39845888);      // 33.5 MB (reused as attn)
  unsigned short* Qs   = (unsigned short*)(ws + 73400320);      // 33.5 MB
  unsigned short* QsT  = (unsigned short*)(ws + 106954752);     // 33.5 MB
  unsigned short* Vt   = (unsigned short*)(ws + 140509184);     // 33.5 MB (transposed: [col][row])
  float*          csum = (float*)         (ws + 174063616);     // 16 KB
  unsigned short* ctxT = (unsigned short*)(ws + 174080000);     // 512 KB
  unsigned short* attn = Qpre;             // Qpre is dead after softmax

  // 0) converts / weight transposes
  cvt_f32_bf16<<<(int)((Sx + 255) / 256), 256, 0, stream>>>(x, Xb, (int)Sx);
  dim3 tgrid(D / 32, D / 32), tblk(32, 8);
  transpose_cvt<<<tgrid, tblk, 0, stream>>>(Wq, WqT, D);
  transpose_cvt<<<tgrid, tblk, 0, stream>>>(Wv, WvT, D);
  transpose_cvt<<<tgrid, tblk, 0, stream>>>(Wo, WoT, D);

  // 1) Q and V projections (V stored transposed for the ctx pass)
  dim3 ggrid(D / 64, M / 128);
  gemm_wmma_bf16<1><<<ggrid, 256, 0, stream>>>(Xb, WqT, bq, Qpre, M, D, D);
  gemm_wmma_bf16<2><<<ggrid, 256, 0, stream>>>(Xb, WvT, bv, Vt,   M, D, D);

  // 2) softmax over d_head, * s  (row-major + transposed outputs)
  const int nseg = M * 16;
  softmax_head_rows<<<nseg / 256, 256, 0, stream>>>(Qpre, Qs, QsT, nseg);

  // 3) column sums of exp(qs) over the sequence dim
  colsum_exp_kernel<<<4096, 256, 0, stream>>>(QsT, csum);

  // 4) ctxT = (s/colsum) * exp(qs)^T @ V  per head (stored transposed)
  ctx_wmma_kernel<<<64, 512, 0, stream>>>(QsT, Vt, csum, ctxT);

  // 5) attn = qs @ ctx, merged-head layout
  attn_wmma_kernel<<<dim3(Nseq / 32, 64), 256, 0, stream>>>(Qs, ctxT, attn);

  // 6) out = attn @ Wo + bo   (f32 output)
  gemm_wmma_bf16<0><<<ggrid, 256, 0, stream>>>(attn, WoT, bo, out, M, D, D);
}
